// CrossMamba_29618094474257
// MI455X (gfx1250) — matrix operand order repaired and checked by reading, not compiled, not run on hardware
//
#include <hip/hip_runtime.h>
#include <hip/hip_bf16.h>

typedef __attribute__((ext_vector_type(16))) _Float16 v16h;
typedef __attribute__((ext_vector_type(8)))  _Float16 v8h;
typedef __attribute__((ext_vector_type(8)))  float    v8f;

constexpr int kB    = 4;
constexpr int kLq   = 1024;
constexpr int kLc   = 1024;
constexpr int kL    = 2048;   // Lc + Lq
constexpr int kDC   = 768;
constexpr int kDM   = 1024;
constexpr int kDS   = 16;
constexpr int kDConv= 4;
constexpr int kDI   = 2048;   // EXP * DM
constexpr int kDtR  = 64;     // DM/16
constexpr int kNXP  = 96;     // DT_RANK + 2*DS
constexpr int kLdsStride = 40; // halves per staged row (padded, 80B, 16B-aligned)

// ---------------------------------------------------------------------------
// Generic WMMA GEMM:  C[m][n] (+)= epilogue( sum_k A[map(m)][k] * W[n][k] )
//   A: (rows x K) fp32, row stride lda, optional per-batch stride, optional
//      row remap: ar = aFlip ? (aFlipLen-1-m) : (aRowOff+m)
//   W: (N x K) fp32 row-major
//   Epilogue: + bias[n] (optional), softplus (act==1), * scale, accumulate.
// Block: 256 threads = 8 waves (4 M x 2 N); block tile 128(M) x 128(N).
// Wave tile: 32(M) x 64(N) = 8 WMMAs per 32-wide K chunk; A frags reused 4x,
// B frags reused 2x. K chunks staged in LDS as f16 (stride 40 halves -> all
// fragment reads are 16B-aligned ds_load_b128 with reduced bank conflicts).
// ---------------------------------------------------------------------------
__global__ __launch_bounds__(256)
void gemm_wmma_f16(const float* __restrict__ A, long long strideA, int lda,
                   int aRowOff, int aFlip, int aFlipLen,
                   const float* __restrict__ W, int ldw,
                   float* __restrict__ C, long long strideC, int ldc,
                   int M, int N, int K,
                   const float* __restrict__ bias, int act, float scale, int accum)
{
    __shared__ __align__(16) _Float16 lA[128 * kLdsStride];
    __shared__ __align__(16) _Float16 lB[128 * kLdsStride];

    const int tid   = threadIdx.x;
    const int lane  = tid & 31;
    const int wv    = tid >> 5;           // wave id 0..7
    const int waveM = wv >> 1;            // 0..3 -> 32-row sub-tile
    const int waveN = wv & 1;             // 0..1 -> 64-col sub-tile
    const int z     = blockIdx.z;
    const int rowBase = blockIdx.y * 128;
    const int colBase = blockIdx.x * 128;

    const float* Ab = A + (long long)z * strideA;
    float*       Cb = C + (long long)z * strideC;

    v8f acc[2][4];
    #pragma unroll
    for (int mi = 0; mi < 2; ++mi)
        #pragma unroll
        for (int nj = 0; nj < 4; ++nj) acc[mi][nj] = v8f{};

    // staging assignment: thread -> (row r in 0..127, k-offset 0 or 16)
    const int r  = tid >> 1;
    const int ko = (tid & 1) * 16;
    const int m_stage = rowBase + r;
    const bool avalid = (m_stage < M);
    const int  ar     = aFlip ? (aFlipLen - 1 - m_stage) : (aRowOff + m_stage);
    const int  wrow   = colBase + r;      // W row == output column
    const bool wvalid = (wrow < N);

    for (int k0 = 0; k0 < K; k0 += 32) {
        // ---- stage A chunk (fp32 -> f16) ----
        {
            _Float16* dst = &lA[r * kLdsStride + ko];
            if (avalid) {
                const float* src = Ab + (long long)ar * lda + (k0 + ko);
                #pragma unroll
                for (int i = 0; i < 16; ++i) dst[i] = (_Float16)src[i];
            } else {
                #pragma unroll
                for (int i = 0; i < 16; ++i) dst[i] = (_Float16)0.0f;
            }
        }
        // ---- stage W chunk (fp32 -> f16) ----
        {
            _Float16* dst = &lB[r * kLdsStride + ko];
            if (wvalid) {
                const float* src = W + (long long)wrow * ldw + (k0 + ko);
                #pragma unroll
                for (int i = 0; i < 16; ++i) dst[i] = (_Float16)src[i];
            } else {
                #pragma unroll
                for (int i = 0; i < 16; ++i) dst[i] = (_Float16)0.0f;
            }
        }
        __syncthreads();

        // ---- 2 A fragments (16x32 each; ISA layout: lane<16 K{0-7,16-23},
        //      lane>=16 K{8-15,24-31}) ----
        const int kbA = (lane < 16) ? 0 : 8;
        v16h afrag[2];
        #pragma unroll
        for (int mi = 0; mi < 2; ++mi) {
            const int arow = waveM * 32 + mi * 16 + (lane & 15);
            const v8h a_lo = *(const v8h*)&lA[arow * kLdsStride + kbA];
            const v8h a_hi = *(const v8h*)&lA[arow * kLdsStride + kbA + 16];
            #pragma unroll
            for (int i = 0; i < 8; ++i) { afrag[mi][i] = a_lo[i]; afrag[mi][i + 8] = a_hi[i]; }
        }

        // ---- 4 B fragments (32x16, ISA layout: lane<16 K{0-7,16-23}, lane>=16 K{8-15,24-31}),
        //      each reused for both A fragments -> 8 WMMAs per chunk ----
        const int kbB = (lane < 16) ? 0 : 8;
        #pragma unroll
        for (int nj = 0; nj < 4; ++nj) {
            const int bcol = waveN * 64 + nj * 16 + (lane & 15);
            const v8h b_lo = *(const v8h*)&lB[bcol * kLdsStride + kbB];
            const v8h b_hi = *(const v8h*)&lB[bcol * kLdsStride + kbB + 16];
            v16h bfrag;
            #pragma unroll
            for (int i = 0; i < 8; ++i) { bfrag[i] = b_lo[i]; bfrag[i + 8] = b_hi[i]; }
            #pragma unroll
            for (int mi = 0; mi < 2; ++mi) {
                acc[mi][nj] = __builtin_amdgcn_wmma_f32_16x16x32_f16(
                                  false, afrag[mi], false, bfrag, (short)0,
                                  acc[mi][nj], false, false);
            }
        }
        __syncthreads();
    }

    // ---- epilogue + store (C/D layout: VGPR e -> M = e + 8*(lane>=16), N = lane&15) ----
    const int rlo  = (lane >= 16) ? 8 : 0;
    const int ccol = lane & 15;
    #pragma unroll
    for (int nj = 0; nj < 4; ++nj) {
        const int gcol = colBase + waveN * 64 + nj * 16 + ccol;
        if (gcol >= N) continue;
        const float bv = bias ? bias[gcol] : 0.0f;
        #pragma unroll
        for (int mi = 0; mi < 2; ++mi) {
            #pragma unroll
            for (int e = 0; e < 8; ++e) {
                const int grow = rowBase + waveM * 32 + mi * 16 + rlo + e;
                if (grow >= M) continue;
                float v = acc[mi][nj][e] + bv;
                if (act == 1) v = (v > 20.0f) ? v : log1pf(__expf(v));   // softplus
                v *= scale;
                float* p = Cb + (long long)grow * ldc + gcol;
                if (accum) *p += v; else *p = v;
            }
        }
    }
}

// x[b, Lc+l, :] = query[b, l, :] + seg_query
__global__ void add_query_kernel(const float* __restrict__ q,
                                 const float* __restrict__ segq,
                                 float* __restrict__ x)
{
    const int idx = blockIdx.x * blockDim.x + threadIdx.x;   // B*Lq*DM threads
    const int d   = idx % kDM;
    const int rest= idx / kDM;
    const int l   = rest % kLq;
    const int b   = rest / kLq;
    x[((long long)b * kL + kLc + l) * kDM + d] = q[idx] + segq[d];
}

// causal depthwise conv (width 4) + bias + SiLU over the u-half of xz
__global__ void conv_silu_kernel(const float* __restrict__ xz,
                                 const float* __restrict__ cw,
                                 const float* __restrict__ cb,
                                 float* __restrict__ u)
{
    const int idx = blockIdx.x * blockDim.x + threadIdx.x;   // B*L*DI threads
    const int d   = idx % kDI;
    const int rest= idx / kDI;
    const int l   = rest % kL;
    const int b   = rest / kL;
    float s = cb[d];
    #pragma unroll
    for (int k = 0; k < kDConv; ++k) {
        const int lk = l - (kDConv - 1) + k;
        if (lk >= 0)
            s += xz[((long long)b * kL + lk) * (2 * kDI) + d] * cw[d * kDConv + k];
    }
    u[idx] = s / (1.0f + __expf(-s));   // SiLU
}

// selective scan: one thread per (b, d) channel, 16 states in registers.
// B_t / C_t (shared across channels of a batch) staged in LDS per 64-step chunk.
// Emits gated output: y = (scan + u*D) * silu(z)
__global__ __launch_bounds__(256)
void scan_kernel(const float* __restrict__ delta,
                 const float* __restrict__ u,
                 const float* __restrict__ xz,       // z gate lives at cols [DI, 2*DI)
                 const float* __restrict__ dbl,      // B_t = cols 64..79, C_t = 80..95
                 const float* __restrict__ A_log,
                 const float* __restrict__ Dv,
                 float* __restrict__ y)
{
    __shared__ float sB[64][kDS];
    __shared__ float sC[64][kDS];
    const int b = blockIdx.y;
    const int d = blockIdx.x * 256 + threadIdx.x;

    float h[kDS];
    float Ar[kDS];
    #pragma unroll
    for (int s = 0; s < kDS; ++s) { h[s] = 0.0f; Ar[s] = -__expf(A_log[d * kDS + s]); }
    const float Dd = Dv[d];
    const long long baseBL = (long long)b * kL;

    for (int l0 = 0; l0 < kL; l0 += 64) {
        for (int t = threadIdx.x; t < 64 * 32; t += 256) {
            const int i = t >> 5, j = t & 31;
            const float v = dbl[(baseBL + l0 + i) * kNXP + kDtR + j];
            if (j < kDS) sB[i][j] = v; else sC[i][j - kDS] = v;
        }
        __syncthreads();
        for (int i = 0; i < 64; ++i) {
            const long long idx = (baseBL + l0 + i) * kDI + d;
            const float dl = delta[idx];
            const float uu = u[idx];
            const float zz = xz[(baseBL + l0 + i) * (2 * kDI) + kDI + d];
            const float du = dl * uu;
            float accv = 0.0f;
            #pragma unroll
            for (int s = 0; s < kDS; ++s) {
                const float dA = __expf(dl * Ar[s]);
                h[s] = h[s] * dA + du * sB[i][s];
                accv += h[s] * sC[i][s];
            }
            const float yy = accv + uu * Dd;
            y[idx] = yy * (zz / (1.0f + __expf(-zz)));
        }
        __syncthreads();
    }
}

extern "C" void kernel_launch(void* const* d_in, const int* in_sizes, int n_in,
                              void* d_out, int out_size, void* d_ws, size_t ws_size,
                              hipStream_t stream)
{
    const float* query     = (const float*)d_in[0];
    const float* context   = (const float*)d_in[1];
    const float* c_in_w    = (const float*)d_in[2];
    const float* seg_c     = (const float*)d_in[3];
    const float* seg_q     = (const float*)d_in[4];
    const float* in_proj_w = (const float*)d_in[5];
    const float* conv_w    = (const float*)d_in[6];
    const float* conv_b    = (const float*)d_in[7];
    const float* x_proj_w  = (const float*)d_in[8];
    const float* dt_proj_w = (const float*)d_in[9];
    const float* dt_proj_b = (const float*)d_in[10];
    const float* A_log     = (const float*)d_in[11];
    const float* Dv        = (const float*)d_in[12];
    const float* out_w     = (const float*)d_in[13];
    float* out = (float*)d_out;

    float* ws = (float*)d_ws;
    size_t off = 0;
    float* x     = ws + off; off += (size_t)kB * kL * kDM;        // 32 MB
    float* xz    = ws + off; off += (size_t)kB * kL * 2 * kDI;    // 128 MB
    float* u     = ws + off; off += (size_t)kB * kL * kDI;        // 64 MB
    float* dbl   = ws + off; off += (size_t)kB * kL * kNXP;       // 3 MB
    float* delta = ws + off; off += (size_t)kB * kL * kDI;        // 64 MB
    float* y     = ws + off; off += (size_t)kB * kL * kDI;        // 64 MB

    const dim3 blk(256);

    // x[:, :Lc, :] = context @ c_in_w^T + seg_context
    gemm_wmma_f16<<<dim3(kDM / 128, kLc / 128, kB), blk, 0, stream>>>(
        context, (long long)kLc * kDC, kDC, 0, 0, 0,
        c_in_w, kDC,
        x, (long long)kL * kDM, kDM,
        kLc, kDM, kDC, seg_c, 0, 1.0f, 0);

    // x[:, Lc:, :] = query + seg_query
    add_query_kernel<<<(kB * kLq * kDM) / 256, 256, 0, stream>>>(query, seg_q, x);

    for (int dir = 0; dir < 2; ++dir) {
        // xz = x(dir) @ in_proj_w^T   (flip rows of A for backward direction)
        gemm_wmma_f16<<<dim3(2 * kDI / 128, kL / 128, kB), blk, 0, stream>>>(
            x, (long long)kL * kDM, kDM, 0, dir, kL,
            in_proj_w, kDM,
            xz, (long long)kL * 2 * kDI, 2 * kDI,
            kL, 2 * kDI, kDM, nullptr, 0, 1.0f, 0);

        // u = silu(causal_conv(xz[:, :, :DI]) + conv_b)
        conv_silu_kernel<<<(kB * kL * kDI) / 256, 256, 0, stream>>>(xz, conv_w, conv_b, u);

        // dbl = u @ x_proj_w^T   (N = 96, bounds-guarded)
        gemm_wmma_f16<<<dim3((kNXP + 127) / 128, (kB * kL) / 128, 1), blk, 0, stream>>>(
            u, 0, kDI, 0, 0, 0,
            x_proj_w, kDI,
            dbl, 0, kNXP,
            kB * kL, kNXP, kDI, nullptr, 0, 1.0f, 0);

        // delta = softplus(dbl[:, :, :64] @ dt_proj_w^T + dt_proj_b)
        gemm_wmma_f16<<<dim3(kDI / 128, (kB * kL) / 128, 1), blk, 0, stream>>>(
            dbl, 0, kNXP, 0, 0, 0,
            dt_proj_w, kDtR,
            delta, 0, kDI,
            kB * kL, kDI, kDtR, dt_proj_b, 1, 1.0f, 0);

        // selective scan + u*D + silu(z) gate -> y (in scan order)
        scan_kernel<<<dim3(kDI / 256, kB), 256, 0, stream>>>(delta, u, xz, dbl, A_log, Dv, y);

        // out += 0.5 * (rows of y mapping to original positions Lc..L-1) @ out_w^T
        //   fwd: scan row Lc+m -> out row m (store)
        //   bwd: scan row Lq-1-m -> out row m (accumulate)
        gemm_wmma_f16<<<dim3(kDM / 128, kLq / 128, kB), blk, 0, stream>>>(
            y, (long long)kL * kDI, kDI, (dir == 0) ? kLc : 0, dir, kLq,
            out_w, kDI,
            out, (long long)kLq * kDM, kDM,
            kLq, kDM, kDI, nullptr, 0, 0.5f, dir);
    }
}
